// HybridGraphConditionedInterventionalPFN_15161234555374
// MI455X (gfx1250) — matrix operand order repaired and checked by reading, not compiled, not run on hardware
//
#include <hip/hip_runtime.h>
#include <hip/hip_fp16.h>
#include <hip/hip_bf16.h>

typedef __attribute__((ext_vector_type(16))) _Float16 v16h;
typedef __attribute__((ext_vector_type(8)))  _Float16 v8h;
typedef __attribute__((ext_vector_type(8)))  float    v8f;
typedef __attribute__((ext_vector_type(4)))  unsigned int u32x4;
typedef __attribute__((ext_vector_type(8)))  int          i32x8;
typedef __attribute__((ext_vector_type(4)))  int          i32x4;

// ---------------- problem constants ----------------
#define B_    4
#define S_    512
#define F_    48
#define FP_   64              // feature length padded to K-multiple of 32
#define D_    256
#define H_    4
#define HD_   64
#define NTR_  384
#define NTE_  128
#define HID_  1024
#define R_    (B_ * S_)        // 2048 feature-attn "rows"
#define T_    (R_ * F_)        // 98304 total tokens
#define NSEQ_ (B_ * F_)        // 192 sample-attn sequences

// ============================================================
// WMMA fragment loaders (wave32, 16x16x32 f16 -> f32). No bounds checks:
// all K are multiples of 32 and rows are 16B aligned by construction.
// ============================================================
__device__ __forceinline__ v16h load_a_frag(const _Float16* __restrict__ A, int lda,
                                            int m0, int k0, int lane) {
  const _Float16* row = A + (long)(m0 + (lane & 15)) * lda + k0 + (lane >> 4) * 16;
  v8h lo = *(const v8h*)(row);
  v8h hi = *(const v8h*)(row + 8);
  return __builtin_shufflevector(lo, hi, 0, 1, 2, 3, 4, 5, 6, 7,
                                 8, 9, 10, 11, 12, 13, 14, 15);
}

__device__ __forceinline__ v16h load_b_frag(const _Float16* __restrict__ Bm, int ldb,
                                            int n0, int k0, int lane) {
  const _Float16* row = Bm + (long)(n0 + (lane & 15)) * ldb + k0 + (lane >> 4) * 16;
  v8h lo = *(const v8h*)(row);
  v8h hi = *(const v8h*)(row + 8);
  return __builtin_shufflevector(lo, hi, 0, 1, 2, 3, 4, 5, 6, 7,
                                 8, 9, 10, 11, 12, 13, 14, 15);
}

// B fragment out of an LDS panel (row-major ROWSxCH chunk, pitch CH halves)
__device__ __forceinline__ v16h load_b_frag_lds(const _Float16* bp, int pitch,
                                                int nloc, int kk, int lane) {
  const _Float16* row = bp + (nloc + (lane & 15)) * pitch + kk + (lane >> 4) * 16;
  v8h lo = *(const v8h*)(row);
  v8h hi = *(const v8h*)(row + 8);
  return __builtin_shufflevector(lo, hi, 0, 1, 2, 3, 4, 5, 6, 7,
                                 8, 9, 10, 11, 12, 13, 14, 15);
}

// ============================================================
// TDM: issue a 2D tile load (rows x cols f16, row stride in elements)
// from global into LDS. D# per CDNA5 ISA ch.8 (group0 + group1, 2D mode).
// This toolchain's builtin takes 6 args (g0, g1, g2, g3, g4, cpol).
// ============================================================
__device__ __forceinline__ void tdm_load_2d(void* lds_dst, const void* gsrc,
                                            unsigned rows, unsigned cols,
                                            unsigned row_stride_elems) {
  unsigned long long ga = (unsigned long long)gsrc;
  unsigned long long la = (unsigned long long)lds_dst;
  u32x4 g0;
  g0[0] = 1u;                                        // count=1, user mode
  g0[1] = (unsigned)la;                              // lds_addr (bytes)
  g0[2] = (unsigned)(ga & 0xffffffffu);              // global_addr[31:0]
  g0[3] = (unsigned)((ga >> 32) & 0x01ffffffu)       // global_addr[56:32]
          | (2u << 30);                              // type = 2 ("image")
  i32x8 g1;
  g1[0] = (int)(1u << 16);                           // data_size=1 (2 bytes)
  g1[1] = (int)((cols & 0xffffu) << 16);             // tensor_dim0[15:0]
  g1[2] = (int)(((cols >> 16) & 0xffffu)             // tensor_dim0[31:16]
          | ((rows & 0xffffu) << 16));               // tensor_dim1[15:0]
  g1[3] = (int)(((rows >> 16) & 0xffffu)             // tensor_dim1[31:16]
          | ((cols & 0xffffu) << 16));               // tile_dim0 = cols
  g1[4] = (int)(rows & 0xffffu);                     // tile_dim1 = rows, tile_dim2=0
  g1[5] = (int)row_stride_elems;                     // tensor_dim0_stride[31:0]
  g1[6] = 0;                                         // stride hi / dim1_stride lo
  g1[7] = 0;
  i32x4 z4 = {0, 0, 0, 0};
  i32x8 z8 = {0, 0, 0, 0, 0, 0, 0, 0};
  __builtin_amdgcn_tensor_load_to_lds(g0, g1, z4, z4, z8, 0);
}

// ============================================================
// Batched WMMA GEMM (direct loads): C = A @ B^T + bias.
// block = 128 threads = 4 waves; wave owns a 16 x (16*NT) C strip.
// ============================================================
template <int NT>
__global__ void k_gemm_wmma(const _Float16* __restrict__ A, int lda, long strideA,
                            const _Float16* __restrict__ Bm, int ldb, long strideB,
                            float* __restrict__ C, int ldc, long strideC,
                            int M, int K, const float* __restrict__ bias) {
  int lane = threadIdx.x & 31;
  int wave = threadIdx.x >> 5;
  int n0 = blockIdx.x * (16 * NT);
  int m0 = (blockIdx.y * 4 + wave) * 16;
  if (m0 >= M) return;
  long bz = blockIdx.z;
  const _Float16* Ab = A  + bz * strideA;
  const _Float16* Bb = Bm + bz * strideB;
  float*          Cb = C  + bz * strideC;

  v8f acc[NT];
#pragma unroll
  for (int t = 0; t < NT; ++t) acc[t] = {};

  for (int k0 = 0; k0 < K; k0 += 32) {
    __builtin_prefetch(Ab + (long)(m0 + (lane & 15)) * lda + k0 + 64, 0, 1);
    v16h a = load_a_frag(Ab, lda, m0, k0, lane);
#pragma unroll
    for (int t = 0; t < NT; ++t) {
      v16h b = load_b_frag(Bb, ldb, n0 + 16 * t, k0, lane);
      acc[t] = __builtin_amdgcn_wmma_f32_16x16x32_f16(false, a, false, b,
                                                      (short)0, acc[t], false, false);
    }
  }

  int mo = (lane >> 4) * 8;
#pragma unroll
  for (int t = 0; t < NT; ++t) {
    int n = n0 + 16 * t + (lane & 15);
    float bv = bias ? bias[n] : 0.0f;
#pragma unroll
    for (int r = 0; r < 8; ++r)
      Cb[(long)(m0 + mo + r) * ldc + n] = acc[t][r] + bv;
  }
}

// ============================================================
// Weight GEMM with TDM-staged B panel in LDS, double buffered.
// Requires: M%64==0, K%128==0, N%64==0, batch==1.
// ============================================================
__global__ void k_gemm_wmma_tdm(const _Float16* __restrict__ A, int lda,
                                const _Float16* __restrict__ Bm, int ldb,
                                float* __restrict__ C, int ldc,
                                int M, int K, const float* __restrict__ bias) {
  constexpr int NT = 4;
  constexpr int CH = 128;                    // K-chunk (elements)
  constexpr int ROWS = 16 * NT;              // 64 B-rows per block
  __shared__ _Float16 bpanel[2][ROWS * CH];  // 2 x 16 KB

  int lane = threadIdx.x & 31;
  int wave = threadIdx.x >> 5;
  int n0 = blockIdx.x * ROWS;
  int m0 = (blockIdx.y * 4 + wave) * 16;
  const _Float16* Bbase = Bm + (long)n0 * ldb;
  int nch = K / CH;

  if (threadIdx.x == 0)
    tdm_load_2d(&bpanel[0][0], Bbase, ROWS, CH, (unsigned)ldb);

  v8f acc[NT];
#pragma unroll
  for (int t = 0; t < NT; ++t) acc[t] = {};

  for (int c = 0; c < nch; ++c) {
    if (threadIdx.x == 0) __builtin_amdgcn_s_wait_tensorcnt(0);
    __syncthreads();                          // panel c is in LDS for all waves
    if (c + 1 < nch && threadIdx.x == 0)
      tdm_load_2d(&bpanel[(c + 1) & 1][0], Bbase + (long)(c + 1) * CH,
                  ROWS, CH, (unsigned)ldb);
    const _Float16* bp = &bpanel[c & 1][0];
#pragma unroll
    for (int kk = 0; kk < CH; kk += 32) {
      int k0 = c * CH + kk;
      v16h a = load_a_frag(A, lda, m0, k0, lane);
#pragma unroll
      for (int t = 0; t < NT; ++t) {
        v16h b = load_b_frag_lds(bp, CH, 16 * t, kk, lane);
        acc[t] = __builtin_amdgcn_wmma_f32_16x16x32_f16(false, a, false, b,
                                                        (short)0, acc[t], false, false);
      }
    }
    __syncthreads();                          // done reading buffer (c&1)
  }

  int mo = (lane >> 4) * 8;
#pragma unroll
  for (int t = 0; t < NT; ++t) {
    int n = n0 + 16 * t + (lane & 15);
    float bv = bias ? bias[n] : 0.0f;
#pragma unroll
    for (int r = 0; r < 8; ++r)
      C[(long)(m0 + mo + r) * ldc + n] = acc[t][r] + bv;
  }
}

// ============================================================
// LayerNorm over D=256, one 256-thread block per token.
// ============================================================
__global__ void k_layernorm(const float* __restrict__ x, long seqStride, long rowOff,
                            const float* __restrict__ w, const float* __restrict__ b,
                            _Float16* __restrict__ out, int Dd) {
  long s = blockIdx.y, i = blockIdx.x;
  const float* in = x + s * seqStride + (rowOff + i) * (long)Dd;
  _Float16* o = out + (s * gridDim.x + i) * (long)Dd;
  int t = threadIdx.x;
  __shared__ float red[256];
  float v = in[t];
  red[t] = v;
  __syncthreads();
  for (int off = 128; off; off >>= 1) { if (t < off) red[t] += red[t + off]; __syncthreads(); }
  float mu = red[0] / Dd;
  __syncthreads();
  float dv = v - mu;
  red[t] = dv * dv;
  __syncthreads();
  for (int off = 128; off; off >>= 1) { if (t < off) red[t] += red[t + off]; __syncthreads(); }
  float rs = rsqrtf(red[0] / Dd + 1e-5f);
  o[t] = (_Float16)(dv * rs * w[t] + b[t]);
}

// f32 token-major (seq*L, H*hd) -> f16 head-major (seq,H,L,hd), optional scale
__global__ void k_headify(const float* __restrict__ src, _Float16* __restrict__ dst,
                          int nseq, int L, int H, int hd, float scale) {
  long n = (long)nseq * L * H * hd;
  for (long idx = blockIdx.x * (long)blockDim.x + threadIdx.x; idx < n;
       idx += (long)gridDim.x * blockDim.x) {
    int d = (int)(idx % hd); long t = idx / hd;
    int i = (int)(t % L);    t /= L;
    int h = (int)(t % H);    long s = t / H;
    float v = src[(s * (long)L + i) * (long)(H * hd) + (long)h * hd + d];
    dst[idx] = (_Float16)(v * scale);
  }
}

// f32 token-major -> f16 transposed heads (seq,H,hd,Lpad), zero-padded i>=L
__global__ void k_headify_t(const float* __restrict__ src, _Float16* __restrict__ dst,
                            int nseq, int L, int Lpad, int H, int hd) {
  long n = (long)nseq * H * hd * Lpad;
  for (long idx = blockIdx.x * (long)blockDim.x + threadIdx.x; idx < n;
       idx += (long)gridDim.x * blockDim.x) {
    int i = (int)(idx % Lpad); long t = idx / Lpad;
    int d = (int)(t % hd);     t /= hd;
    int h = (int)(t % H);      long s = t / H;
    float v = (i < L) ? src[(s * (long)L + i) * (long)(H * hd) + (long)h * hd + d] : 0.0f;
    dst[idx] = (_Float16)v;
  }
}

// f32 head-major (seq,H,L,hd) -> f16 token-major (seq*L, H*hd)
__global__ void k_mergeheads(const float* __restrict__ src, _Float16* __restrict__ dst,
                             int nseq, int L, int H, int hd) {
  long n = (long)nseq * L * H * hd;
  for (long idx = blockIdx.x * (long)blockDim.x + threadIdx.x; idx < n;
       idx += (long)gridDim.x * blockDim.x) {
    int d = (int)(idx % hd); long t = idx / hd;
    int h = (int)(t % H);    t /= H;
    int i = (int)(t % L);    long s = t / L;
    dst[idx] = (_Float16)src[((s * (long)H + h) * L + i) * (long)hd + d];
  }
}

// plain row softmax, one wave per row, f16 output with pitch ldp (pad zeroed)
__global__ void k_softmax(const float* __restrict__ s, _Float16* __restrict__ p,
                          int ncols, int ldp) {
  long row = blockIdx.x;
  const float* sr = s + row * (long)ncols;
  _Float16* pr = p + row * (long)ldp;
  int lane = threadIdx.x;
  float mx = -3.0e38f;
  for (int c = lane; c < ncols; c += 32) mx = fmaxf(mx, sr[c]);
  for (int o = 16; o; o >>= 1) mx = fmaxf(mx, __shfl_xor(mx, o));
  float sum = 0.f;
  for (int c = lane; c < ncols; c += 32) sum += __expf(sr[c] - mx);
  for (int o = 16; o; o >>= 1) sum += __shfl_xor(sum, o);
  float inv = 1.0f / sum;
  for (int c = lane; c < ldp; c += 32)
    pr[c] = (c < ncols) ? (_Float16)(__expf(sr[c] - mx) * inv) : (_Float16)0.0f;
}

// feature-attention softmax with graph mask on heads < 2; pitch-64 f16 output
__global__ void k_softmax_feat(const float* __restrict__ s, _Float16* __restrict__ p,
                               const int* __restrict__ mask, int F, int Fpad,
                               int H, int Srows) {
  long row = blockIdx.x;                 // ((r*H + h)*F + i)
  int i = (int)(row % F); long t = row / F;
  int h = (int)(t % H);   long r = t / H;
  int b = (int)(r / Srows);
  const float* sr = s + row * (long)F;
  _Float16* pr = p + row * (long)Fpad;
  const int* mrow = mask + ((long)b * F + i) * F;
  bool useMask = (h < 2);
  int lane = threadIdx.x;
  float mx = -3.0e38f;
  for (int c = lane; c < F; c += 32) {
    float v = (useMask && mrow[c] != 1) ? -3.0e38f : sr[c];
    mx = fmaxf(mx, v);
  }
  for (int o = 16; o; o >>= 1) mx = fmaxf(mx, __shfl_xor(mx, o));
  float sum = 0.f;
  for (int c = lane; c < F; c += 32) {
    float v = (useMask && mrow[c] != 1) ? -3.0e38f : sr[c];
    sum += __expf(v - mx);
  }
  for (int o = 16; o; o >>= 1) sum += __shfl_xor(sum, o);
  float inv = 1.0f / sum;
  for (int c = lane; c < Fpad; c += 32) {
    if (c < F) {
      float v = (useMask && mrow[c] != 1) ? -3.0e38f : sr[c];
      pr[c] = (_Float16)(__expf(v - mx) * inv);
    } else {
      pr[c] = (_Float16)0.0f;
    }
  }
}

// (B,S,F,D) <-> (B,F,S,D). one block per destination token, 256 threads copy D
__global__ void k_permute(const float* __restrict__ src, float* __restrict__ dst,
                          int Bq, int Sq, int Fq, int Dd, int toCol) {
  long tok = blockIdx.x;
  int t = threadIdx.x;
  if (toCol) {
    long s = tok % Sq; long f = (tok / Sq) % Fq; long b = tok / ((long)Sq * Fq);
    dst[tok * Dd + t] = src[((b * Sq + s) * Fq + f) * (long)Dd + t];
  } else {
    long f = tok % Fq; long s = (tok / Fq) % Sq; long b = tok / ((long)Sq * Fq);
    dst[tok * Dd + t] = src[((b * Fq + f) * Sq + s) * (long)Dd + t];
  }
}

__global__ void k_add(const float* __restrict__ a, const float* __restrict__ b,
                      float* __restrict__ c, long n) {
  for (long i = blockIdx.x * (long)blockDim.x + threadIdx.x; i < n;
       i += (long)gridDim.x * blockDim.x)
    c[i] = a[i] + b[i];
}

__global__ void k_add_strided(float* __restrict__ dst, const float* __restrict__ src,
                              int L, int Dd, long seqStride, long rowOff) {
  long s = blockIdx.y, i = blockIdx.x;
  int t = threadIdx.x;
  dst[s * seqStride + (rowOff + i) * (long)Dd + t] += src[(s * (long)L + i) * Dd + t];
}

__global__ void k_cvt16(const float* __restrict__ a, _Float16* __restrict__ o, long n) {
  for (long i = blockIdx.x * (long)blockDim.x + threadIdx.x; i < n;
       i += (long)gridDim.x * blockDim.x)
    o[i] = (_Float16)a[i];
}

__global__ void k_silu_mul(const float* __restrict__ h, const float* __restrict__ g,
                           _Float16* __restrict__ o, long n) {
  for (long i = blockIdx.x * (long)blockDim.x + threadIdx.x; i < n;
       i += (long)gridDim.x * blockDim.x) {
    float gv = g[i];
    o[i] = (_Float16)(h[i] * (gv / (1.0f + __expf(-gv))));
  }
}

// ============================================================
// host orchestration
// ============================================================
static void gemm(hipStream_t st,
                 const _Float16* A, int lda, long sA,
                 const _Float16* Bm, int ldb, long sB,
                 float* C, int ldc, long sC,
                 int M, int N, int K, int batch, const float* bias) {
  if (batch == 1 && (M % 64 == 0) && (N % 64 == 0) && (K % 128 == 0)) {
    dim3 grid((unsigned)(N / 64), (unsigned)(M / 64), 1);
    k_gemm_wmma_tdm<<<grid, 128, 0, st>>>(A, lda, Bm, ldb, C, ldc, M, K, bias);
  } else if (N % 64 == 0) {
    dim3 grid((unsigned)(N / 64), (unsigned)((M + 63) / 64), (unsigned)batch);
    k_gemm_wmma<4><<<grid, 128, 0, st>>>(A, lda, sA, Bm, ldb, sB, C, ldc, sC, M, K, bias);
  } else {  // N % 48 == 0 (feature dimension)
    dim3 grid((unsigned)(N / 48), (unsigned)((M + 63) / 64), (unsigned)batch);
    k_gemm_wmma<3><<<grid, 128, 0, st>>>(A, lda, sA, Bm, ldb, sB, C, ldc, sC, M, K, bias);
  }
}

extern "C" void kernel_launch(void* const* d_in, const int* in_sizes, int n_in,
                              void* d_out, int out_size, void* d_ws, size_t ws_size,
                              hipStream_t stream) {
  (void)in_sizes; (void)n_in; (void)out_size; (void)ws_size;
  const float* x_in  = (const float*)d_in[0];
  const int*   mask  = (const int*)d_in[1];
  const float* fW[4] = {(const float*)d_in[4],  (const float*)d_in[5],
                        (const float*)d_in[6],  (const float*)d_in[7]};
  const float* fB[4] = {(const float*)d_in[8],  (const float*)d_in[9],
                        (const float*)d_in[10], (const float*)d_in[11]};
  const float* tW[4] = {(const float*)d_in[12], (const float*)d_in[13],
                        (const float*)d_in[14], (const float*)d_in[15]};
  const float* tB[4] = {(const float*)d_in[16], (const float*)d_in[17],
                        (const float*)d_in[18], (const float*)d_in[19]};
  const float* eW[4] = {(const float*)d_in[20], (const float*)d_in[21],
                        (const float*)d_in[22], (const float*)d_in[23]};
  const float* eB[4] = {(const float*)d_in[24], (const float*)d_in[25],
                        (const float*)d_in[26], (const float*)d_in[27]};
  const float* ln_feat_w = (const float*)d_in[28]; const float* ln_feat_b = (const float*)d_in[29];
  const float* ln_tr_w   = (const float*)d_in[30]; const float* ln_tr_b   = (const float*)d_in[31];
  const float* ln_te_w   = (const float*)d_in[32]; const float* ln_te_b   = (const float*)d_in[33];
  const float* ln_mlp_w  = (const float*)d_in[34]; const float* ln_mlp_b  = (const float*)d_in[35];
  const float* fc1_w = (const float*)d_in[36]; const float* fc1_b = (const float*)d_in[37];
  const float* gate_w = (const float*)d_in[38]; const float* gate_b = (const float*)d_in[39];
  const float* fc2_w = (const float*)d_in[40]; const float* fc2_b = (const float*)d_in[41];
  float* out = (float*)d_out;

  char* cur = (char*)d_ws;
  auto alloc = [&](size_t bytes) -> void* {
    void* r = (void*)cur;
    cur += (bytes + 255) & ~(size_t)255;
    return r;
  };
  const long TD = (long)T_ * D_;
  const long SCORES_MAX = (long)NSEQ_ * H_ * NTR_ * NTR_;
  const long VT_MAX     = (long)R_ * H_ * HD_ * FP_;

  _Float16* w16_fW[4]; _Float16* w16_tW[4]; _Float16* w16_eW[4];
  for (int i = 0; i < 4; ++i) w16_fW[i] = (_Float16*)alloc((size_t)D_ * D_ * 2);
  for (int i = 0; i < 4; ++i) w16_tW[i] = (_Float16*)alloc((size_t)D_ * D_ * 2);
  for (int i = 0; i < 4; ++i) w16_eW[i] = (_Float16*)alloc((size_t)D_ * D_ * 2);
  _Float16* w16_fc1  = (_Float16*)alloc((size_t)HID_ * D_ * 2);
  _Float16* w16_gate = (_Float16*)alloc((size_t)HID_ * D_ * 2);
  _Float16* w16_fc2  = (_Float16*)alloc((size_t)D_ * HID_ * 2);

  _Float16* xn16   = (_Float16*)alloc((size_t)TD * 2);
  _Float16* qh16   = (_Float16*)alloc((size_t)TD * 2);
  _Float16* kh16   = (_Float16*)alloc((size_t)TD * 2);
  _Float16* vt16   = (_Float16*)alloc((size_t)VT_MAX * 2);
  _Float16* o16    = (_Float16*)alloc((size_t)TD * 2);
  _Float16* prob16 = (_Float16*)alloc((size_t)SCORES_MAX * 2);
  float* bufX    = (float*)alloc((size_t)TD * 4);
  float* bufC    = (float*)alloc((size_t)TD * 4);
  float* bufT    = (float*)alloc((size_t)TD * 4);
  float* scores  = (float*)alloc((size_t)SCORES_MAX * 4);
  float* hid32a  = (float*)alloc((size_t)T_ * HID_ * 4);
  float* hid32b  = (float*)alloc((size_t)T_ * HID_ * 4);
  _Float16* hid16 = (_Float16*)alloc((size_t)T_ * HID_ * 2);

  const int GS = 4096, BS = 256;
  for (int i = 0; i < 4; ++i) {
    k_cvt16<<<GS, BS, 0, stream>>>(fW[i], w16_fW[i], (long)D_ * D_);
    k_cvt16<<<GS, BS, 0, stream>>>(tW[i], w16_tW[i], (long)D_ * D_);
    k_cvt16<<<GS, BS, 0, stream>>>(eW[i], w16_eW[i], (long)D_ * D_);
  }
  k_cvt16<<<GS, BS, 0, stream>>>(fc1_w,  w16_fc1,  (long)HID_ * D_);
  k_cvt16<<<GS, BS, 0, stream>>>(gate_w, w16_gate, (long)HID_ * D_);
  k_cvt16<<<GS, BS, 0, stream>>>(fc2_w,  w16_fc2,  (long)D_ * HID_);

  const float qscale = 0.125f;  // 1/sqrt(64)

  // =============== Stage A: feature attention ===============
  {
    dim3 g((unsigned)T_, 1);
    k_layernorm<<<g, 256, 0, stream>>>(x_in, 0, 0, ln_feat_w, ln_feat_b, xn16, D_);
    gemm(stream, xn16, D_, 0, w16_fW[0], D_, 0, bufT, D_, 0, T_, D_, D_, 1, fB[0]);
    k_headify<<<GS, BS, 0, stream>>>(bufT, qh16, R_, F_, H_, HD_, qscale);
    gemm(stream, xn16, D_, 0, w16_fW[1], D_, 0, bufT, D_, 0, T_, D_, D_, 1, fB[1]);
    k_headify<<<GS, BS, 0, stream>>>(bufT, kh16, R_, F_, H_, HD_, 1.0f);
    gemm(stream, xn16, D_, 0, w16_fW[2], D_, 0, bufT, D_, 0, T_, D_, D_, 1, fB[2]);
    k_headify_t<<<GS, BS, 0, stream>>>(bufT, vt16, R_, F_, FP_, H_, HD_);
    gemm(stream, qh16, HD_, (long)F_ * HD_, kh16, HD_, (long)F_ * HD_,
         scores, F_, (long)F_ * F_, F_, F_, HD_, R_ * H_, nullptr);
    k_softmax_feat<<<(unsigned)(R_ * H_ * F_), 32, 0, stream>>>(scores, prob16, mask,
                                                                F_, FP_, H_, S_);
    gemm(stream, prob16, FP_, (long)F_ * FP_, vt16, FP_, (long)HD_ * FP_,
         bufT, HD_, (long)F_ * HD_, F_, HD_, FP_, R_ * H_, nullptr);
    k_mergeheads<<<GS, BS, 0, stream>>>(bufT, o16, R_, F_, H_, HD_);
    gemm(stream, o16, D_, 0, w16_fW[3], D_, 0, bufT, D_, 0, T_, D_, D_, 1, fB[3]);
    k_add<<<GS, BS, 0, stream>>>(x_in, bufT, bufX, TD);
  }

  // =============== Stage B: sample attention (column view) ===============
  k_permute<<<(unsigned)T_, 256, 0, stream>>>(bufX, bufC, B_, S_, F_, D_, 1);
  { // train self-attention
    const int Mtr = NSEQ_ * NTR_;
    dim3 g((unsigned)NTR_, (unsigned)NSEQ_);
    k_layernorm<<<g, 256, 0, stream>>>(bufC, (long)S_ * D_, 0, ln_tr_w, ln_tr_b, xn16, D_);
    gemm(stream, xn16, D_, 0, w16_tW[0], D_, 0, bufT, D_, 0, Mtr, D_, D_, 1, tB[0]);
    k_headify<<<GS, BS, 0, stream>>>(bufT, qh16, NSEQ_, NTR_, H_, HD_, qscale);
    gemm(stream, xn16, D_, 0, w16_tW[1], D_, 0, bufT, D_, 0, Mtr, D_, D_, 1, tB[1]);
    k_headify<<<GS, BS, 0, stream>>>(bufT, kh16, NSEQ_, NTR_, H_, HD_, 1.0f);
    gemm(stream, xn16, D_, 0, w16_tW[2], D_, 0, bufT, D_, 0, Mtr, D_, D_, 1, tB[2]);
    k_headify_t<<<GS, BS, 0, stream>>>(bufT, vt16, NSEQ_, NTR_, NTR_, H_, HD_);
    gemm(stream, qh16, HD_, (long)NTR_ * HD_, kh16, HD_, (long)NTR_ * HD_,
         scores, NTR_, (long)NTR_ * NTR_, NTR_, NTR_, HD_, NSEQ_ * H_, nullptr);
    k_softmax<<<(unsigned)(NSEQ_ * H_ * NTR_), 32, 0, stream>>>(scores, prob16, NTR_, NTR_);
    gemm(stream, prob16, NTR_, (long)NTR_ * NTR_, vt16, NTR_, (long)HD_ * NTR_,
         bufT, HD_, (long)NTR_ * HD_, NTR_, HD_, NTR_, NSEQ_ * H_, nullptr);
    k_mergeheads<<<GS, BS, 0, stream>>>(bufT, o16, NSEQ_, NTR_, H_, HD_);
    gemm(stream, o16, D_, 0, w16_tW[3], D_, 0, bufT, D_, 0, Mtr, D_, D_, 1, tB[3]);
    k_add_strided<<<g, 256, 0, stream>>>(bufC, bufT, NTR_, D_, (long)S_ * D_, 0);
  }
  { // test cross-attention
    const int Mte = NSEQ_ * NTE_, Mtr = NSEQ_ * NTR_;
    _Float16* qn16 = xn16;
    _Float16* kv16 = xn16 + (long)Mte * D_;
    dim3 gq((unsigned)NTE_, (unsigned)NSEQ_), gk((unsigned)NTR_, (unsigned)NSEQ_);
    k_layernorm<<<gq, 256, 0, stream>>>(bufC, (long)S_ * D_, NTR_, ln_te_w, ln_te_b, qn16, D_);
    k_layernorm<<<gk, 256, 0, stream>>>(bufC, (long)S_ * D_, 0,    ln_te_w, ln_te_b, kv16, D_);
    gemm(stream, qn16, D_, 0, w16_eW[0], D_, 0, bufT, D_, 0, Mte, D_, D_, 1, eB[0]);
    k_headify<<<GS, BS, 0, stream>>>(bufT, qh16, NSEQ_, NTE_, H_, HD_, qscale);
    gemm(stream, kv16, D_, 0, w16_eW[1], D_, 0, bufT, D_, 0, Mtr, D_, D_, 1, eB[1]);
    k_headify<<<GS, BS, 0, stream>>>(bufT, kh16, NSEQ_, NTR_, H_, HD_, 1.0f);
    gemm(stream, kv16, D_, 0, w16_eW[2], D_, 0, bufT, D_, 0, Mtr, D_, D_, 1, eB[2]);
    k_headify_t<<<GS, BS, 0, stream>>>(bufT, vt16, NSEQ_, NTR_, NTR_, H_, HD_);
    gemm(stream, qh16, HD_, (long)NTE_ * HD_, kh16, HD_, (long)NTR_ * HD_,
         scores, NTR_, (long)NTE_ * NTR_, NTE_, NTR_, HD_, NSEQ_ * H_, nullptr);
    k_softmax<<<(unsigned)(NSEQ_ * H_ * NTE_), 32, 0, stream>>>(scores, prob16, NTR_, NTR_);
    gemm(stream, prob16, NTR_, (long)NTE_ * NTR_, vt16, NTR_, (long)HD_ * NTR_,
         bufT, HD_, (long)NTE_ * HD_, NTE_, HD_, NTR_, NSEQ_ * H_, nullptr);
    k_mergeheads<<<GS, BS, 0, stream>>>(bufT, o16, NSEQ_, NTE_, H_, HD_);
    gemm(stream, o16, D_, 0, w16_eW[3], D_, 0, bufT, D_, 0, Mte, D_, D_, 1, eB[3]);
    k_add_strided<<<gq, 256, 0, stream>>>(bufC, bufT, NTE_, D_, (long)S_ * D_, NTR_);
  }

  // =============== Stage C: SwiGLU MLP ===============
  k_permute<<<(unsigned)T_, 256, 0, stream>>>(bufC, bufX, B_, S_, F_, D_, 0);
  {
    dim3 g((unsigned)T_, 1);
    k_layernorm<<<g, 256, 0, stream>>>(bufX, 0, 0, ln_mlp_w, ln_mlp_b, xn16, D_);
    gemm(stream, xn16, D_, 0, w16_fc1,  D_, 0, hid32a, HID_, 0, T_, HID_, D_, 1, fc1_b);
    gemm(stream, xn16, D_, 0, w16_gate, D_, 0, hid32b, HID_, 0, T_, HID_, D_, 1, gate_b);
    k_silu_mul<<<GS, BS, 0, stream>>>(hid32a, hid32b, hid16, (long)T_ * HID_);
    gemm(stream, hid16, HID_, 0, w16_fc2, HID_, 0, bufT, D_, 0, T_, D_, HID_, 1, fc2_b);
    k_add<<<GS, BS, 0, stream>>>(bufX, bufT, out, TD);
  }
}
